// NetVLAD_26946624815577
// MI455X (gfx1250) — matrix operand order repaired and checked by reading, not compiled, not run on hardware
//
#include <hip/hip_runtime.h>
#include <hip/hip_bf16.h>
#include <math.h>

// ---------------------------------------------------------------------------
// NetVLAD fused kernel for gfx1250 (MI455X):
//   wave32 + V_WMMA_F32_16X16X32_F16 + GLOBAL_LOAD_ASYNC_TO_LDS_B128
// ---------------------------------------------------------------------------

typedef _Float16 half8_t __attribute__((ext_vector_type(8)));
typedef _Float16 v16h    __attribute__((ext_vector_type(16)));
typedef float    v8f     __attribute__((ext_vector_type(8)));

#define N_B    64
#define C_CH   512
#define P_PIX  1600
#define K_CL   64
#define TILE   32
#define SPLITS 4           // 256 blocks total; tiles per split: 13,13,12,12
#define EPS_N  1e-12f

// LDS layout (bytes). Pitches keep WMMA fragment runs 16B-aligned and spread
// lane address strides across banks.
#define WPITCH 520         // halves: rows of w_lds and xn2 (512 used + pad)
#define X1P    40          // halves: xn1 row pitch (32 used + pad)
#define AP     40          // halves: a_kp row pitch
#define LP     72          // floats: logitsT row pitch (64 used + pad)

#define OFF_W     0
#define OFF_XN1   (OFF_W    + K_CL * WPITCH * 2)     // 66560
#define OFF_XN2   (OFF_XN1  + C_CH * X1P * 2)        // +40960
#define OFF_PART  (OFF_XN2  + TILE * WPITCH * 2)     // +33280
#define OFF_LOGT  (OFF_PART + 16 * 8 * 32 * 4)       // +16384
#define OFF_AKP   (OFF_LOGT + TILE * LP * 4)         // +9216
#define OFF_NORMP (OFF_AKP  + K_CL * AP * 2)         // +5120
#define OFF_INVN  (OFF_NORMP + 16 * 32 * 4)          // +2048
#define OFF_ASUM  (OFF_INVN + 32 * 4)                // +128
#define OFF_STG   (OFF_ASUM + 64 * 4)                // x-tile stage, double buf
#define STG_BYTES (C_CH * TILE * 4)                  // 65536 per buffer
#define SMEM_TOTAL (OFF_STG + 2 * STG_BYTES)         // = 305024 bytes

// One v_wmma_f32_16x16x32_f16: A/B fragments are two 16-byte LDS runs each.
__device__ __forceinline__ v8f wmma16(const _Float16* a0, const _Float16* a1,
                                      const _Float16* b0, const _Float16* b1,
                                      v8f c) {
  union U { v16h v; half8_t h[2]; };
  U ua, ub;
  ua.h[0] = *(const half8_t*)a0;
  ua.h[1] = *(const half8_t*)a1;
  ub.h[0] = *(const half8_t*)b0;
  ub.h[1] = *(const half8_t*)b1;
  return __builtin_amdgcn_wmma_f32_16x16x32_f16(
      /*neg_a=*/false, ua.v, /*neg_b=*/false, ub.v,
      /*c_mod=*/(short)0, c, /*reuse_a=*/false, /*reuse_b=*/false);
}

// ---------------------------------------------------------------------------
// Kernel 1: per (n, pixel-split): async-stage x tile to LDS, normalize,
// logits (WMMA), softmax, VLAD partial (WMMA accumulated in 64 VGPRs/lane),
// atomic combine into d_out.
// ---------------------------------------------------------------------------
__global__ __launch_bounds__(512) void netvlad_accum(
    const float* __restrict__ x, const float* __restrict__ conv_w,
    float* __restrict__ vlad_out, float* __restrict__ asum_g)
{
  extern __shared__ __align__(16) char smem[];
  _Float16* w_lds = (_Float16*)(smem + OFF_W);     // [64][WPITCH]  (k-major)
  _Float16* xn1   = (_Float16*)(smem + OFF_XN1);   // [512][X1P]   (c-major)
  _Float16* xn2   = (_Float16*)(smem + OFF_XN2);   // [32][WPITCH] (p-major)
  float*    part  = (float*)(smem + OFF_PART);     // [16][8][32]
  float*    logT  = (float*)(smem + OFF_LOGT);     // [32][LP]  logits^T
  _Float16* a_kp  = (_Float16*)(smem + OFF_AKP);   // [64][AP]
  float*    normp = (float*)(smem + OFF_NORMP);    // [16][32]
  float*    invn  = (float*)(smem + OFF_INVN);     // [32]
  float*    asum  = (float*)(smem + OFF_ASUM);     // [64]
  float*    stg   = (float*)(smem + OFF_STG);      // [2][512][32]

  const int tid = threadIdx.x;
  const int wv  = tid >> 5;        // wave 0..15
  const int l   = tid & 31;        // lane
  const int l16 = l & 15;
  const int g   = l >> 4;          // half-wave group

  const int split = blockIdx.x & (SPLITS - 1);
  const int n     = blockIdx.x >> 2;
  const int tstart = (split < 2) ? split * 13 : 26 + (split - 2) * 12;
  const int titers = (split < 2) ? 13 : 12;

  const unsigned nbase = (unsigned)n * (C_CH * P_PIX * 4u);  // byte offset of n

  // Issue 8 async b128 copies per thread: one 512x32 f32 tile -> LDS stage.
  auto issue_stage = [&](int tile, int buf) {
    const unsigned p0b = (unsigned)tile * (TILE * 4u);
    #pragma unroll
    for (int q = 0; q < 8; ++q) {
      int ch  = tid + 512 * q;             // chunk 0..4095
      int c   = ch >> 3;                   // channel row
      int sub = (ch & 7) * 16;             // byte offset within 128B row
      unsigned lds_off = (unsigned)OFF_STG + (unsigned)buf * STG_BYTES +
                         (unsigned)c * (TILE * 4u) + (unsigned)sub;
      unsigned goff = nbase + (unsigned)c * (P_PIX * 4u) + p0b + (unsigned)sub;
      asm volatile("global_load_async_to_lds_b128 %0, %1, %2 offset:0"
                   :: "v"(lds_off), "v"(goff), "s"(x)
                   : "memory");
    }
  };

  // conv_w (K=64, C=512) -> f16 LDS, [k][c]
  for (int i = 0; i < 64; ++i) {
    int idx = i * 512 + tid;
    w_lds[(idx >> 9) * WPITCH + (idx & 511)] = (_Float16)conv_w[idx];
  }
  if (tid < 64) asum[tid] = 0.0f;

  issue_stage(tstart, 0);                  // prefill buffer 0

  // persistent VLAD accumulators: wave wv owns k-tile (wv&3), c-tiles (wv>>2)*8..+7
  v8f acc[8];
  v8f vzero = {};
  #pragma unroll
  for (int j = 0; j < 8; ++j) acc[j] = vzero;

  float asum_priv[4] = {0.f, 0.f, 0.f, 0.f};

  const int lp = tid & 31;                 // pixel within tile
  const int cs = tid >> 5;                 // channel-pair slice 0..15

  for (int it = 0; it < titers; ++it) {
    const int buf = it & 1;
    if (it + 1 < titers) {
      issue_stage(tstart + it + 1, buf ^ 1);   // overlap next tile with compute
      asm volatile("s_wait_asynccnt 8" ::: "memory");  // current buffer done
    } else {
      asm volatile("s_wait_asynccnt 0" ::: "memory");
    }
    __syncthreads();               // stage visible; xn/a free from prev readers

    const float* sb = stg + buf * (STG_BYTES / 4) + lp;   // column lp

    // -------- phase 1a: per-pixel sumsq from LDS stage ---------------------
    float ss = 0.f;
    #pragma unroll
    for (int i = 0; i < 16; ++i) {
      int c = i * 32 + cs * 2;
      float v0 = sb[c * TILE];
      float v1 = sb[(c + 1) * TILE];
      ss += v0 * v0 + v1 * v1;
    }
    normp[cs * 32 + lp] = ss;
    __syncthreads();
    if (tid < 32) {
      float s = 0.f;
      #pragma unroll
      for (int i = 0; i < 16; ++i) s += normp[i * 32 + tid];
      invn[tid] = 1.0f / fmaxf(sqrtf(s), EPS_N);
    }
    __syncthreads();

    // -------- phase 1b: normalize -> f16 xn1 [c][p], xn2 [p][c] ------------
    {
      float inv = invn[lp];
      #pragma unroll
      for (int i = 0; i < 16; ++i) {
        int c = i * 32 + cs * 2;
        float v0 = sb[c * TILE];
        float v1 = sb[(c + 1) * TILE];
        _Float16 h0 = (_Float16)(v0 * inv);
        _Float16 h1 = (_Float16)(v1 * inv);
        xn1[c * X1P + lp] = h0;
        xn1[(c + 1) * X1P + lp] = h1;
        union { _Float16 h[2]; unsigned u; } pk;
        pk.h[0] = h0; pk.h[1] = h1;
        *(unsigned*)&xn2[lp * WPITCH + c] = pk.u;   // c even -> 4B aligned
      }
    }
    __syncthreads();

    // -------- phase 2: logits = W(64x512) * xn(512x32), K split 2x ---------
    {
      int t   = wv & 7;
      int kt1 = t >> 1, pt = t & 1, kh = wv >> 3;
      v8f acc1 = vzero;
      const _Float16* wrow = w_lds + (kt1 * 16 + l16) * WPITCH;
      const _Float16* prow = xn2 + (pt * 16 + l16) * WPITCH;
      #pragma unroll
      for (int s = 0; s < 8; ++s) {
        int c0 = kh * 256 + s * 32;
        acc1 = wmma16(wrow + c0 + g * 8, wrow + c0 + 16 + g * 8,
                      prow + c0 + g * 8, prow + c0 + 16 + g * 8, acc1);
      }
      #pragma unroll
      for (int r = 0; r < 8; ++r) part[(wv * 8 + r) * 32 + l] = acc1[r];
    }
    __syncthreads();

    // -------- phase 3: combine K-halves, transpose to logT[p][k] -----------
    #pragma unroll
    for (int q = 0; q < 4; ++q) {
      int e  = tid + 512 * q;        // 2048 elements = 8 tiles * 16 * 16
      int t  = e >> 8;
      int M  = (e >> 4) & 15;
      int Nn = e & 15;
      int ln = (M >> 3) * 16 + Nn;
      int rr = M & 7;
      float v = part[(t * 8 + rr) * 32 + ln] + part[((t + 8) * 8 + rr) * 32 + ln];
      logT[((t & 1) * 16 + Nn) * LP + ((t >> 1) * 16 + M)] = v;
    }
    __syncthreads();

    // -------- phase 4: softmax over K=64 per pixel (16 threads/pixel) ------
    {
      const int sp = tid >> 4;
      const int sl = tid & 15;
      float v0 = logT[sp * LP + sl];
      float v1 = logT[sp * LP + sl + 16];
      float v2 = logT[sp * LP + sl + 32];
      float v3 = logT[sp * LP + sl + 48];
      float m = fmaxf(fmaxf(v0, v1), fmaxf(v2, v3));
      for (int d = 8; d >= 1; d >>= 1) m = fmaxf(m, __shfl_xor(m, d, 16));
      float e0 = __expf(v0 - m), e1 = __expf(v1 - m);
      float e2 = __expf(v2 - m), e3 = __expf(v3 - m);
      float s = e0 + e1 + e2 + e3;
      for (int d = 8; d >= 1; d >>= 1) s += __shfl_xor(s, d, 16);
      float inv = 1.0f / s;
      e0 *= inv; e1 *= inv; e2 *= inv; e3 *= inv;
      a_kp[(sl)      * AP + sp] = (_Float16)e0;
      a_kp[(sl + 16) * AP + sp] = (_Float16)e1;
      a_kp[(sl + 32) * AP + sp] = (_Float16)e2;
      a_kp[(sl + 48) * AP + sp] = (_Float16)e3;
      asum_priv[0] += e0; asum_priv[1] += e1;
      asum_priv[2] += e2; asum_priv[3] += e3;
    }
    __syncthreads();

    // -------- phase 5: VLAD partial V += a(64x32) * xn^T(32x512) -----------
    {
      int kt = wv & 3, cg = wv >> 2;
      const _Float16* arow = a_kp + (kt * 16 + l16) * AP;
      const _Float16* a0 = arow + g * 8;
      const _Float16* a1 = arow + 16 + g * 8;
      #pragma unroll
      for (int j = 0; j < 8; ++j) {
        int c = (cg * 8 + j) * 16 + l16;
        const _Float16* brow = xn1 + c * X1P;
        acc[j] = wmma16(a0, a1, brow + g * 8, brow + 16 + g * 8, acc[j]);
      }
    }
  }

  // -------- epilogue: combine asum + atomic-add partial V into d_out -------
  __syncthreads();
  {
    const int sl = tid & 15;
    atomicAdd(&asum[sl],      asum_priv[0]);
    atomicAdd(&asum[sl + 16], asum_priv[1]);
    atomicAdd(&asum[sl + 32], asum_priv[2]);
    atomicAdd(&asum[sl + 48], asum_priv[3]);
  }
  __syncthreads();
  if (tid < 64) atomicAdd(&asum_g[n * 64 + tid], asum[tid]);
  {
    int kt = wv & 3, cg = wv >> 2;
    #pragma unroll
    for (int j = 0; j < 8; ++j) {
      int c = (cg * 8 + j) * 16 + l16;
      #pragma unroll
      for (int r = 0; r < 8; ++r) {
        int k = kt * 16 + r + 8 * g;     // C/D layout: M = r + 8*g, N = l16
        atomicAdd(&vlad_out[((size_t)n * K_CL + k) * C_CH + c], acc[j][r]);
      }
    }
  }
}

// ---------------------------------------------------------------------------
// Kernel 2: finalize per n: subtract asum*centroid, intra-L2 over C,
// global L2 over K*C; in-place on d_out. One block per n, 512 threads.
// ---------------------------------------------------------------------------
__global__ __launch_bounds__(512) void netvlad_finalize(
    const float* __restrict__ centroids, const float* __restrict__ asum_g,
    float* __restrict__ out)
{
  __shared__ float asum_sh[64];
  __shared__ float ksq[64];
  __shared__ float invk[64];
  __shared__ float invG_sh;
  const int tid = threadIdx.x;
  const int n   = blockIdx.x;
  if (tid < 64) asum_sh[tid] = asum_g[n * 64 + tid];
  __syncthreads();

  const int kk    = tid >> 3;          // 8 threads per k row
  const int cbase = (tid & 7) * 64;    // 64 channels each
  float* vrow = out + (size_t)n * (K_CL * C_CH) + kk * C_CH + cbase;
  const float* crow = centroids + kk * C_CH + cbase;
  const float as = asum_sh[kk];

  float ss = 0.f;
  for (int j = 0; j < 64; ++j) {
    float v = vrow[j] - as * crow[j];
    ss += v * v;
  }
  ss += __shfl_xor(ss, 4, 8);
  ss += __shfl_xor(ss, 2, 8);
  ss += __shfl_xor(ss, 1, 8);
  if ((tid & 7) == 0) ksq[kk] = ss;
  __syncthreads();

  if (tid < 64) invk[tid] = 1.0f / fmaxf(sqrtf(ksq[tid]), EPS_N);
  __syncthreads();
  if (tid == 0) {
    float G = 0.f;
    for (int i = 0; i < 64; ++i) G += ksq[i] * invk[i] * invk[i];
    invG_sh = 1.0f / fmaxf(sqrtf(G), EPS_N);
  }
  __syncthreads();

  const float scale = invk[kk] * invG_sh;
  for (int j = 0; j < 64; ++j) {
    float v = vrow[j] - as * crow[j];
    vrow[j] = v * scale;
  }
}

// ---------------------------------------------------------------------------
extern "C" void kernel_launch(void* const* d_in, const int* in_sizes, int n_in,
                              void* d_out, int out_size, void* d_ws, size_t ws_size,
                              hipStream_t stream) {
  const float* x         = (const float*)d_in[0];  // [64,512,40,40]
  const float* conv_w    = (const float*)d_in[1];  // [64,512,1,1]
  const float* centroids = (const float*)d_in[2];  // [64,512]
  float* out    = (float*)d_out;                   // [64, 64*512]
  float* asum_g = (float*)d_ws;                    // [64,64] f32 = 16 KB

  hipMemsetAsync(d_out, 0, (size_t)N_B * K_CL * C_CH * sizeof(float), stream);
  hipMemsetAsync(d_ws, 0, (size_t)N_B * K_CL * sizeof(float), stream);

  hipLaunchKernelGGL(netvlad_accum, dim3(N_B * SPLITS), dim3(512), SMEM_TOTAL,
                     stream, x, conv_w, out, asum_g);
  hipLaunchKernelGGL(netvlad_finalize, dim3(N_B), dim3(512), 0,
                     stream, centroids, asum_g, out);
}
